// TemporalGAT_55705725829930
// MI455X (gfx1250) — matrix operand order repaired and checked by reading, not compiled, not run on hardware
//
#include <hip/hip_runtime.h>
#include <hip/hip_bf16.h>
#include <hip/hip_fp16.h>

// ---------------- problem constants ----------------
#define NN     50000
#define FIN    128
#define HDIM   128
#define NHEADS 4
#define CDIM   32
#define NE     800000

typedef __attribute__((ext_vector_type(16))) _Float16 v16h;
typedef __attribute__((ext_vector_type(8)))  float    v8f;

// ---------------- WMMA fragment helpers (CDNA5 16x16x32 f16 layouts) --------
// A (16x32 f16): lane L<16 holds row L, K in {0..15}; lane>=16: K in {16..31}
// Per lane the 16 K values are contiguous -> compiler emits ds_load_b128 x2.
__device__ __forceinline__ v16h frag_a_lds(const _Float16* A, int lda, int kbase) {
  int lane = threadIdx.x & 31;
  int row  = lane & 15;
  int kg   = (lane >> 4) << 4;             // 0 or 16
  v16h a;
#pragma unroll
  for (int i = 0; i < 16; ++i) {
    int k = kbase + kg + i;
    a[i] = A[row * lda + k];
  }
  return a;
}
// B fragment from pre-packed (swizzled) weights: each fragment is 32 lanes x 16
// halves stored contiguously; per-lane 32B aligned -> two global_load_b128.
__device__ __forceinline__ v16h frag_b_pk(const _Float16* Bp, int nk, int ntile, int kstep) {
  int lane = threadIdx.x & 31;
  const _Float16* p = Bp + ((((size_t)ntile * nk + kstep) * 32 + lane) << 4);
  return *(const v16h*)p;
}

__device__ __forceinline__ float wave_sum(float v) {
#pragma unroll
  for (int m = 16; m > 0; m >>= 1) v += __shfl_xor(v, m, 32);
  return v;
}

// float -> order-preserving unsigned encoding (for atomicMax on floats)
__device__ __forceinline__ unsigned encf(float x) {
  unsigned u = __float_as_uint(x);
  return (u & 0x80000000u) ? ~u : (u | 0x80000000u);
}
__device__ __forceinline__ float decf(unsigned e) {
  return (e & 0x80000000u) ? __uint_as_float(e & 0x7fffffffu) : __uint_as_float(~e);
}

__device__ __forceinline__ float lrelu02(float x) { return x > 0.0f ? x : 0.2f * x; }

// ---------------- small utility kernels ----------------
// Pack row-major f32 weight W[K][N] into per-fragment WMMA-B layout (f16):
// out[((ntile*nk + kstep)*32 + lane)*16 + i] = W[kstep*32 + (lane>>4)*16 + i][ntile*16 + (lane&15)]
__global__ void k_pack_b(const float* __restrict__ W, _Float16* __restrict__ out, int K, int N) {
  int t = blockIdx.x * 256 + threadIdx.x;
  if (t >= K * N) return;
  int nk    = K >> 5;
  int i     = t & 15;
  int lane  = (t >> 4) & 31;
  int fr    = t >> 9;               // ntile*nk + kstep
  int kstep = fr % nk;
  int ntile = fr / nk;
  int k = (kstep << 5) + ((lane >> 4) << 4) + i;
  int n = (ntile << 4) + (lane & 15);
  out[t] = (_Float16)W[k * N + n];
}
__global__ void k_fill0(float* __restrict__ p, int n) {
  int t = blockIdx.x * 256 + threadIdx.x;
  if (t < n) p[t] = 0.0f;
}
// v[k*4+hd] = sum_c W_edge[k*128 + hd*32 + c] * att_edge[hd*32 + c], both layers
__global__ void k_make_v(const float* __restrict__ ew1, const float* __restrict__ ae1,
                         const float* __restrict__ ew2, const float* __restrict__ ae2,
                         float* __restrict__ v1, float* __restrict__ v2) {
  int t = blockIdx.x * 256 + threadIdx.x;
  if (t >= 1024) return;
  int l = t >> 9, r = t & 511, k = r >> 2, hd = r & 3;
  const float* W = l ? ew2 : ew1;
  const float* a = l ? ae2 : ae1;
  float* v = l ? v2 : v1;
  float s = 0.0f;
#pragma unroll
  for (int c = 0; c < CDIM; ++c) s += W[k * HDIM + hd * CDIM + c] * a[hd * CDIM + c];
  v[k * 4 + hd] = s;
}

// ---------------- K1: h1 = LN(relu(x @ ip_w1 + b1))  (f16 out, [N,256]) -----
__global__ __launch_bounds__(512) void k_proj1(const float* __restrict__ x,
    const _Float16* __restrict__ w1p, const float* __restrict__ b1,
    const float* __restrict__ lnw, const float* __restrict__ lnb,
    _Float16* __restrict__ h1h) {
  __shared__ _Float16 As[16][128];
  __shared__ float    Cs[16][256];
  int tid = threadIdx.x, lane = tid & 31, wave = tid >> 5;
  int rb = blockIdx.x * 16;                        // N % 16 == 0
  for (int idx = tid; idx < 16 * 128; idx += 512) {
    int r = idx >> 7, c = idx & 127;
    As[r][c] = (_Float16)x[(size_t)(rb + r) * 128 + c];
  }
  __syncthreads();
  int nb = wave * 16;                              // 16 waves -> 256 cols
  v8f acc = {};
#pragma unroll
  for (int ks = 0; ks < 4; ++ks) {
    v16h a = frag_a_lds(&As[0][0], 128, ks * 32);
    v16h b = frag_b_pk(w1p, 4, wave, ks);
    acc = __builtin_amdgcn_wmma_f32_16x16x32_f16(false, a, false, b, (short)0, acc, false, false);
  }
  int col = nb + (lane & 15);
  int rg  = (lane >> 4) << 3;
#pragma unroll
  for (int r = 0; r < 8; ++r)
    Cs[rg + r][col] = fmaxf(acc[r] + b1[col], 0.0f);   // relu BEFORE LN
  __syncthreads();
  int row = wave;                                  // one wave per row
  float s = 0.0f, q = 0.0f;
#pragma unroll
  for (int i = 0; i < 8; ++i) { float v = Cs[row][lane * 8 + i]; s += v; q += v * v; }
  s = wave_sum(s); q = wave_sum(q);
  float mean = s * (1.0f / 256.0f);
  float var  = q * (1.0f / 256.0f) - mean * mean;
  float rstd = rsqrtf(var + 1e-5f);
#pragma unroll
  for (int i = 0; i < 8; ++i) {
    int j = lane * 8 + i;
    float v = (Cs[row][j] - mean) * rstd * lnw[j] + lnb[j];
    h1h[(size_t)(rb + row) * 256 + j] = (_Float16)v;
  }
}

// -------- K2: feat = clip(relu(LN(h1 @ ip_w2 + b2)), -10, 10)  (f32+f16) ----
__global__ __launch_bounds__(256) void k_proj2(const _Float16* __restrict__ h1h,
    const _Float16* __restrict__ w2p, const float* __restrict__ b2,
    const float* __restrict__ pnw, const float* __restrict__ pnb,
    float* __restrict__ feat, _Float16* __restrict__ feath) {
  __shared__ _Float16 As[16][256];
  __shared__ float    Cs[16][128];
  int tid = threadIdx.x, lane = tid & 31, wave = tid >> 5;
  int rb = blockIdx.x * 16;
  for (int idx = tid; idx < 16 * 256; idx += 256) {
    int r = idx >> 8, c = idx & 255;
    As[r][c] = h1h[(size_t)(rb + r) * 256 + c];
  }
  __syncthreads();
  int nb = wave * 16;
  v8f acc = {};
#pragma unroll
  for (int ks = 0; ks < 8; ++ks) {
    v16h a = frag_a_lds(&As[0][0], 256, ks * 32);
    v16h b = frag_b_pk(w2p, 8, wave, ks);
    acc = __builtin_amdgcn_wmma_f32_16x16x32_f16(false, a, false, b, (short)0, acc, false, false);
  }
  int col = nb + (lane & 15);
  int rg  = (lane >> 4) << 3;
#pragma unroll
  for (int r = 0; r < 8; ++r) Cs[rg + r][col] = acc[r] + b2[col];
  __syncthreads();
#pragma unroll
  for (int rr = 0; rr < 2; ++rr) {                 // 8 waves x 2 rows = 16 rows
    int row = wave + rr * 8;
    float s = 0.0f, q = 0.0f;
#pragma unroll
    for (int i = 0; i < 4; ++i) { float v = Cs[row][lane * 4 + i]; s += v; q += v * v; }
    s = wave_sum(s); q = wave_sum(q);
    float mean = s * (1.0f / 128.0f);
    float rstd = rsqrtf(q * (1.0f / 128.0f) - mean * mean + 1e-5f);
#pragma unroll
    for (int i = 0; i < 4; ++i) {
      int j = lane * 4 + i;
      float v = (Cs[row][j] - mean) * rstd * pnw[j] + pnb[j];
      v = fminf(fmaxf(v, 0.0f), 10.0f);            // relu then clip
      feat[(size_t)(rb + row) * 128 + j]  = v;
      feath[(size_t)(rb + row) * 128 + j] = (_Float16)v;
    }
  }
}

// -- K3: fused temporal MLP + a_edge = tw @ v (both layers) + loop sums ------
__global__ __launch_bounds__(256) void k_temporal(const float* __restrict__ et,
    const int* __restrict__ dst,
    const float* __restrict__ taw1, const float* __restrict__ tab1,
    const float* __restrict__ talnw, const float* __restrict__ talnb,
    const _Float16* __restrict__ taw2p, const float* __restrict__ tab2,
    const float* __restrict__ v1, const float* __restrict__ v2,
    float* __restrict__ ae1, float* __restrict__ ae2,
    float* __restrict__ lsum1, float* __restrict__ lsum2, float* __restrict__ deg) {
  __shared__ float    Tm[16][128];
  __shared__ _Float16 Th[16][128];
  __shared__ float    Out[16][128];
  int tid = threadIdx.x, lane = tid & 31, wave = tid >> 5;
  long long eb = (long long)blockIdx.x * 16;       // E % 16 == 0
  for (int idx = tid; idx < 2048; idx += 256) {
    int r = idx >> 7, c = idx & 127;
    Tm[r][c] = et[eb + r] * taw1[c] + tab1[c];
  }
  __syncthreads();
#pragma unroll
  for (int rr = 0; rr < 2; ++rr) {                 // LN then relu -> f16
    int row = wave + rr * 8;
    float s = 0.0f, q = 0.0f;
#pragma unroll
    for (int i = 0; i < 4; ++i) { float v = Tm[row][lane * 4 + i]; s += v; q += v * v; }
    s = wave_sum(s); q = wave_sum(q);
    float mean = s * (1.0f / 128.0f);
    float rstd = rsqrtf(q * (1.0f / 128.0f) - mean * mean + 1e-5f);
#pragma unroll
    for (int i = 0; i < 4; ++i) {
      int j = lane * 4 + i;
      float v = (Tm[row][j] - mean) * rstd * talnw[j] + talnb[j];
      Th[row][j] = (_Float16)fmaxf(v, 0.0f);
    }
  }
  __syncthreads();
  int nb = wave * 16;
  v8f acc = {};
#pragma unroll
  for (int ks = 0; ks < 4; ++ks) {
    v16h a = frag_a_lds(&Th[0][0], 128, ks * 32);
    v16h b = frag_b_pk(taw2p, 4, wave, ks);
    acc = __builtin_amdgcn_wmma_f32_16x16x32_f16(false, a, false, b, (short)0, acc, false, false);
  }
  int col = nb + (lane & 15);
  int rg  = (lane >> 4) << 3;
#pragma unroll
  for (int r = 0; r < 8; ++r) {
    float v = tanhf(acc[r] + tab2[col]);
    Out[rg + r][col] = fminf(fmaxf(v, -3.0f), 3.0f);
  }
  __syncthreads();
  if (tid < 64) {
    int i = tid >> 2, hd = tid & 3;
    float s1 = 0.0f, s2 = 0.0f;
#pragma unroll 8
    for (int k = 0; k < 128; ++k) {
      float t = Out[i][k];
      s1 += t * v1[k * 4 + hd];
      s2 += t * v2[k * 4 + hd];
    }
    long long e = eb + i;
    int d = dst[e];
    ae1[e * 4 + hd] = s1;
    ae2[e * 4 + hd] = s2;
    atomicAdd(&lsum1[(size_t)d * 4 + hd], s1);
    atomicAdd(&lsum2[(size_t)d * 4 + hd], s2);
    if (hd == 0) atomicAdd(&deg[d], 1.0f);
  }
}

// ---- K4: hW = feat @ W (WMMA) + fused a_src/a_dst head dots ---------------
__global__ __launch_bounds__(256) void k_hw(const _Float16* __restrict__ feath,
    const _Float16* __restrict__ wp,
    const float* __restrict__ attsrc, const float* __restrict__ attdst,
    float* __restrict__ hW, float* __restrict__ asrc, float* __restrict__ adst) {
  __shared__ _Float16 As[16][128];
  __shared__ float    Cs[16][128];
  int tid = threadIdx.x, lane = tid & 31, wave = tid >> 5;
  int rb = blockIdx.x * 16;
  for (int idx = tid; idx < 2048; idx += 256) {
    int r = idx >> 7, c = idx & 127;
    As[r][c] = feath[(size_t)(rb + r) * 128 + c];
  }
  __syncthreads();
  int nb = wave * 16;
  v8f acc = {};
#pragma unroll
  for (int ks = 0; ks < 4; ++ks) {
    v16h a = frag_a_lds(&As[0][0], 128, ks * 32);
    v16h b = frag_b_pk(wp, 4, wave, ks);
    acc = __builtin_amdgcn_wmma_f32_16x16x32_f16(false, a, false, b, (short)0, acc, false, false);
  }
  int col = nb + (lane & 15);
  int rg  = (lane >> 4) << 3;
#pragma unroll
  for (int r = 0; r < 8; ++r) {
    float v = acc[r];
    Cs[rg + r][col] = v;
    hW[(size_t)(rb + rg + r) * 128 + col] = v;
  }
  __syncthreads();
  if (tid < 64) {
    int i = tid >> 2, hd = tid & 3;
    float s1 = 0.0f, s2 = 0.0f;
#pragma unroll
    for (int c = 0; c < CDIM; ++c) {
      float t = Cs[i][hd * CDIM + c];
      s1 += t * attsrc[hd * CDIM + c];
      s2 += t * attdst[hd * CDIM + c];
    }
    asrc[(size_t)(rb + i) * 4 + hd] = s1;
    adst[(size_t)(rb + i) * 4 + hd] = s2;
  }
}

// ---- K5: init segment-max with self-loop alpha ----------------------------
__global__ void k_minit(const float* __restrict__ asrc, const float* __restrict__ adst,
                        const float* __restrict__ lsum, const float* __restrict__ deg,
                        unsigned* __restrict__ menc) {
  int t = blockIdx.x * 256 + threadIdx.x;
  if (t >= NN * 4) return;
  int n = t >> 2;
  float al = lrelu02(asrc[t] + adst[t] + lsum[t] / fmaxf(deg[n], 1.0f));
  menc[t] = encf(al);
}

// ---- K6: per-edge alpha + atomic segment max ------------------------------
__global__ void k_edge_max(const int* __restrict__ src, const int* __restrict__ dst,
                           const float* __restrict__ asrc, const float* __restrict__ adst,
                           const float* __restrict__ ae, float* __restrict__ aa,
                           unsigned* __restrict__ menc) {
  long long t = (long long)blockIdx.x * 256 + threadIdx.x;
  if (t >= (long long)NE * 4) return;
  long long e = t >> 2; int hd = t & 3;
  int s = src[e], d = dst[e];
  float a = lrelu02(asrc[(size_t)s * 4 + hd] + adst[(size_t)d * 4 + hd] + ae[t]);
  aa[t] = a;
  atomicMax(&menc[(size_t)d * 4 + hd], encf(a));
}

// ---- K7: init sum + out_raw with self-loop contribution -------------------
__global__ void k_sinit(const float* __restrict__ asrc, const float* __restrict__ adst,
                        const float* __restrict__ lsum, const float* __restrict__ deg,
                        const unsigned* __restrict__ menc, const float* __restrict__ hW,
                        float* __restrict__ ssum, float* __restrict__ oraw) {
  long long t = (long long)blockIdx.x * 256 + threadIdx.x;
  if (t >= (long long)NN * 128) return;
  int n = (int)(t >> 7), j = (int)(t & 127), hd = j >> 5;
  float al = lrelu02(asrc[(size_t)n * 4 + hd] + adst[(size_t)n * 4 + hd] +
                     lsum[(size_t)n * 4 + hd] / fmaxf(deg[n], 1.0f));
  float p = expf(al - decf(menc[(size_t)n * 4 + hd]));
  oraw[t] = p * hW[t];
  if ((j & 31) == 0) ssum[(size_t)n * 4 + hd] = p;
}

// ---- K8: edge aggregation: p=exp(aa-m[dst]); s+=p; oraw[dst]+=p*hW[src] ----
__global__ __launch_bounds__(256) void k_edge_agg(const int* __restrict__ src,
    const int* __restrict__ dst, const float* __restrict__ aa,
    const unsigned* __restrict__ menc, const float* __restrict__ hW,
    float* __restrict__ ssum, float* __restrict__ oraw) {
  long long t = (long long)blockIdx.x * 256 + threadIdx.x;
  long long e = t >> 7;
  if (e >= NE) return;
  int j = (int)(t & 127), hd = j >> 5;
  int s = src[e], d = dst[e];
  float p = expf(aa[e * 4 + hd] - decf(menc[(size_t)d * 4 + hd]));
  if ((j & 31) == 0) atomicAdd(&ssum[(size_t)d * 4 + hd], p);
  atomicAdd(&oraw[(size_t)d * 128 + j], p * hW[(size_t)s * 128 + j]);
}

// ---- K9: node epilogue: /s + bias + residual, clip, LN, ELU ---------------
__global__ __launch_bounds__(128) void k_epilogue(const float* __restrict__ oraw,
    const float* __restrict__ ssum, const float* __restrict__ bias,
    const float* __restrict__ res, const float* __restrict__ nw,
    const float* __restrict__ nb_, float* __restrict__ outF, _Float16* __restrict__ outH) {
  __shared__ float red[128];
  int n = blockIdx.x, j = threadIdx.x, hd = j >> 5;
  float y = oraw[(size_t)n * 128 + j] / (ssum[(size_t)n * 4 + hd] + 1e-16f) +
            bias[j] + res[(size_t)n * 128 + j];
  y = fminf(fmaxf(y, -10.0f), 10.0f);
  red[j] = y; __syncthreads();
  for (int st = 64; st > 0; st >>= 1) { if (j < st) red[j] += red[j + st]; __syncthreads(); }
  float mean = red[0] * (1.0f / 128.0f); __syncthreads();
  float dyy = (y - mean) * (y - mean);
  red[j] = dyy; __syncthreads();
  for (int st = 64; st > 0; st >>= 1) { if (j < st) red[j] += red[j + st]; __syncthreads(); }
  float var = red[0] * (1.0f / 128.0f);
  float v = (y - mean) * rsqrtf(var + 1e-5f) * nw[j] + nb_[j];
  v = v > 0.0f ? v : expm1f(v);                    // ELU
  outF[(size_t)n * 128 + j] = v;
  if (outH) outH[(size_t)n * 128 + j] = (_Float16)v;
}

// ---------------- host-side orchestration ----------------
extern "C" void kernel_launch(void* const* d_in, const int* in_sizes, int n_in,
                              void* d_out, int out_size, void* d_ws, size_t ws_size,
                              hipStream_t stream) {
  const float* x       = (const float*)d_in[0];
  const float* et      = (const float*)d_in[1];
  const float* ip_w1   = (const float*)d_in[2];
  const float* ip_b1   = (const float*)d_in[3];
  const float* ip_lnw  = (const float*)d_in[4];
  const float* ip_lnb  = (const float*)d_in[5];
  const float* ip_w2   = (const float*)d_in[6];
  const float* ip_b2   = (const float*)d_in[7];
  const float* pn_w    = (const float*)d_in[8];
  const float* pn_b    = (const float*)d_in[9];
  const float* ta_w1   = (const float*)d_in[10];
  const float* ta_b1   = (const float*)d_in[11];
  const float* ta_lnw  = (const float*)d_in[12];
  const float* ta_lnb  = (const float*)d_in[13];
  const float* ta_w2   = (const float*)d_in[14];
  const float* ta_b2   = (const float*)d_in[15];
  const float* g1_w    = (const float*)d_in[16];
  const float* g1_as   = (const float*)d_in[17];
  const float* g1_ad   = (const float*)d_in[18];
  const float* g1_ew   = (const float*)d_in[19];
  const float* g1_ae   = (const float*)d_in[20];
  const float* g1_b    = (const float*)d_in[21];
  const float* g2_w    = (const float*)d_in[22];
  const float* g2_as   = (const float*)d_in[23];
  const float* g2_ad   = (const float*)d_in[24];
  const float* g2_ew   = (const float*)d_in[25];
  const float* g2_ae   = (const float*)d_in[26];
  const float* g2_b    = (const float*)d_in[27];
  const float* n1_w    = (const float*)d_in[28];
  const float* n1_b    = (const float*)d_in[29];
  const float* n2_w    = (const float*)d_in[30];
  const float* n2_b    = (const float*)d_in[31];
  const int*   eidx    = (const int*)d_in[32];
  const int*   src     = eidx;
  const int*   dst     = eidx + NE;
  float* out = (float*)d_out;

  char* base = (char*)d_ws;
  size_t off = 0;
  auto alloc = [&](size_t bytes) -> void* {
    void* p = base + off;
    off += (bytes + 255) & ~(size_t)255;
    return p;
  };
  _Float16* w1p    = (_Float16*)alloc(128 * 256 * 2);
  _Float16* w2p    = (_Float16*)alloc(256 * 128 * 2);
  _Float16* taw2p  = (_Float16*)alloc(128 * 128 * 2);
  _Float16* g1wp   = (_Float16*)alloc(128 * 128 * 2);
  _Float16* g2wp   = (_Float16*)alloc(128 * 128 * 2);
  float*    v1     = (float*)alloc(128 * 4 * 4);
  float*    v2     = (float*)alloc(128 * 4 * 4);
  _Float16* h1h    = (_Float16*)alloc((size_t)NN * 256 * 2);
  float*    feat1  = (float*)alloc((size_t)NN * 128 * 4);
  _Float16* feat1h = (_Float16*)alloc((size_t)NN * 128 * 2);
  float*    feat2  = (float*)alloc((size_t)NN * 128 * 4);
  _Float16* feat2h = (_Float16*)alloc((size_t)NN * 128 * 2);
  float*    hW     = (float*)alloc((size_t)NN * 128 * 4);
  float*    asrc   = (float*)alloc((size_t)NN * 4 * 4);
  float*    adst   = (float*)alloc((size_t)NN * 4 * 4);
  float*    ae1    = (float*)alloc((size_t)NE * 4 * 4);
  float*    ae2    = (float*)alloc((size_t)NE * 4 * 4);
  float*    lsum1  = (float*)alloc((size_t)NN * 4 * 4);
  float*    lsum2  = (float*)alloc((size_t)NN * 4 * 4);
  float*    deg    = (float*)alloc((size_t)NN * 4);
  float*    aa     = (float*)alloc((size_t)NE * 4 * 4);
  unsigned* menc   = (unsigned*)alloc((size_t)NN * 4 * 4);
  float*    ssum   = (float*)alloc((size_t)NN * 4 * 4);
  float*    oraw   = (float*)alloc((size_t)NN * 128 * 4);
  (void)ws_size; (void)n_in; (void)in_sizes; (void)out_size;

  // weight prep: pack into WMMA-B fragment layout (one-time, tiny)
  k_pack_b<<<128, 256, 0, stream>>>(ip_w1, w1p, 128, 256);
  k_pack_b<<<128, 256, 0, stream>>>(ip_w2, w2p, 256, 128);
  k_pack_b<<<64, 256, 0, stream>>>(ta_w2, taw2p, 128, 128);
  k_pack_b<<<64, 256, 0, stream>>>(g1_w, g1wp, 128, 128);
  k_pack_b<<<64, 256, 0, stream>>>(g2_w, g2wp, 128, 128);
  k_make_v<<<4, 256, 0, stream>>>(g1_ew, g1_ae, g2_ew, g2_ae, v1, v2);
  k_fill0<<<(NN * 4 + 255) / 256, 256, 0, stream>>>(lsum1, NN * 4);
  k_fill0<<<(NN * 4 + 255) / 256, 256, 0, stream>>>(lsum2, NN * 4);
  k_fill0<<<(NN + 255) / 256, 256, 0, stream>>>(deg, NN);

  // input projection
  k_proj1<<<NN / 16, 512, 0, stream>>>(x, w1p, ip_b1, ip_lnw, ip_lnb, h1h);
  k_proj2<<<NN / 16, 256, 0, stream>>>(h1h, w2p, ip_b2, pn_w, pn_b, feat1, feat1h);

  // temporal MLP fused with a_edge for both layers + loop-attr sums
  k_temporal<<<NE / 16, 256, 0, stream>>>(et, dst, ta_w1, ta_b1, ta_lnw, ta_lnb,
                                          taw2p, ta_b2, v1, v2, ae1, ae2,
                                          lsum1, lsum2, deg);

  // ---- GAT layer 1 ----
  k_hw<<<NN / 16, 256, 0, stream>>>(feat1h, g1wp, g1_as, g1_ad, hW, asrc, adst);
  k_minit<<<(NN * 4 + 255) / 256, 256, 0, stream>>>(asrc, adst, lsum1, deg, menc);
  k_edge_max<<<(NE * 4) / 256, 256, 0, stream>>>(src, dst, asrc, adst, ae1, aa, menc);
  k_sinit<<<(NN * 128) / 256, 256, 0, stream>>>(asrc, adst, lsum1, deg, menc, hW, ssum, oraw);
  k_edge_agg<<<((long long)NE * 128) / 256, 256, 0, stream>>>(src, dst, aa, menc, hW, ssum, oraw);
  k_epilogue<<<NN, 128, 0, stream>>>(oraw, ssum, g1_b, feat1, n1_w, n1_b, feat2, feat2h);

  // ---- GAT layer 2 ----
  k_hw<<<NN / 16, 256, 0, stream>>>(feat2h, g2wp, g2_as, g2_ad, hW, asrc, adst);
  k_minit<<<(NN * 4 + 255) / 256, 256, 0, stream>>>(asrc, adst, lsum2, deg, menc);
  k_edge_max<<<(NE * 4) / 256, 256, 0, stream>>>(src, dst, asrc, adst, ae2, aa, menc);
  k_sinit<<<(NN * 128) / 256, 256, 0, stream>>>(asrc, adst, lsum2, deg, menc, hW, ssum, oraw);
  k_edge_agg<<<((long long)NE * 128) / 256, 256, 0, stream>>>(src, dst, aa, menc, hW, ssum, oraw);
  k_epilogue<<<NN, 128, 0, stream>>>(oraw, ssum, g2_b, feat2, n2_w, n2_b, out, (_Float16*)nullptr);
}
